// MoELayerSim_343597384113
// MI455X (gfx1250) — compile-verified, matrix operand order checked by reading, not run on hardware
//
#include <hip/hip_runtime.h>
#include <stdint.h>

// ---------------------------------------------------------------------------
// MoE (top-2 of 8 experts), H=1024, F=4096, T=4096 tokens.
// Route first, then bf16 WMMA GEMMs only on routed (expert,token) pairs.
// Weights are pre-transformed once into a fragment-native tiled layout so the
// GEMM hot loop has zero pack VALU; A and B tiles stream to LDS with
// global_load_async_to_lds_b128 (ASYNCcnt) overlapped with WMMA compute.
// Wave tile 32x128 (2 A-frags x 8 B-frags -> 16 WMMA, B reused 2x).
// ---------------------------------------------------------------------------

constexpr int B_ = 2, S_ = 2048, H_ = 1024, F_ = 4096, E_ = 8;
constexpr int T_  = B_ * S_;     // 4096 tokens
constexpr int RT_ = T_ * 2;      // max routed slots (top-2)

constexpr int TM = 128;  // token tile (4 wave rows x 32)
constexpr int TN = 256;  // col tile (2 wave cols x 128)
constexpr int TK = 32;   // K step (one bf16 WMMA depth)
constexpr int LDK = 40;  // padded K stride in LDS (80 B, 16B multiple)

typedef __attribute__((ext_vector_type(16))) __bf16 v16bf;
typedef __attribute__((ext_vector_type(8)))  float  v8f;

#define DI static __device__ __forceinline__

DI unsigned short f2bf(float f) {        // f32 -> bf16, round-to-nearest-even
    uint32_t u = __builtin_bit_cast(uint32_t, f);
    u = (u + 0x7FFFu + ((u >> 16) & 1u)) >> 16;
    return (unsigned short)u;
}

DI v16bf mk_frag(uint4 lo, uint4 hi) {   // elems 0..7 = lo bytes, 8..15 = hi
    v16bf a;
    __builtin_memcpy(&a, &lo, 16);
    __builtin_memcpy((char*)&a + 16, &hi, 16);
    return a;
}

// CDNA5 async DMA: copy 16B global -> LDS, tracked by ASYNCcnt.
// VDST = LDS byte address (low 32 bits of generic pointer), VADDR = global.
DI void async_copy16(const void* gsrc, void* ldst) {
    asm volatile("global_load_async_to_lds_b128 %0, %1, off"
                 :: "v"((uint32_t)(uintptr_t)ldst), "v"(gsrc) : "memory");
}
DI void wait_async0() { asm volatile("s_wait_asynccnt 0x0" ::: "memory"); }

// --------------------------- workspace layout ------------------------------
constexpr size_t XB_OFF  = 0;                                   // x bf16 [T][H]
constexpr size_t W1B_OFF = XB_OFF  + (size_t)T_ * H_ * 2;       // w1t tiled bf16
constexpr size_t W2B_OFF = W1B_OFF + (size_t)E_ * H_ * F_ * 2;  // w2t tiled bf16
constexpr size_t HID_OFF = W2B_OFF + (size_t)E_ * F_ * H_ * 2;  // hidden bf16 [RT][F]
constexpr size_t TOK_OFF = HID_OFF + (size_t)RT_ * F_ * 2;      // int [E][T]
constexpr size_t PRB_OFF = TOK_OFF + (size_t)E_ * T_ * 4;       // float [E][T]
constexpr size_t CNT_OFF = PRB_OFF + (size_t)E_ * T_ * 4;       // int [E]
constexpr size_t BAS_OFF = CNT_OFF + 64;                        // int [E]
constexpr size_t PSM_OFF = BAS_OFF + 64;                        // float [E]
constexpr size_t ZB_OFF  = PSM_OFF + 64;                        // 256B zeros

// --------------------------- init / convert --------------------------------
__global__ void moe_zero(float* __restrict__ out, int* __restrict__ counts,
                         float* __restrict__ probsum, uint32_t* __restrict__ zbuf) {
    size_t i  = (size_t)blockIdx.x * blockDim.x + threadIdx.x;
    size_t st = (size_t)gridDim.x * blockDim.x;
    for (size_t j = i; j < (size_t)T_ * H_; j += st) out[j] = 0.f;
    if (i < E_) { counts[i] = 0; probsum[i] = 0.f; }
    if (i < 64) zbuf[i] = 0u;
}

__global__ void moe_cvt(const float4* __restrict__ src,
                        ushort4* __restrict__ dst, size_t n4) {
    size_t i  = (size_t)blockIdx.x * blockDim.x + threadIdx.x;
    size_t st = (size_t)gridDim.x * blockDim.x;
    for (; i < n4; i += st) {
        float4 v = src[i];
        ushort4 r;
        r.x = f2bf(v.x); r.y = f2bf(v.y); r.z = f2bf(v.z); r.w = f2bf(v.w);
        dst[i] = r;
    }
}

// Weight transform: w[e][k][n] f32 -> wt[(e*KT + k/32)*ND + n][32] bf16,
// i.e. per (ktile, col) the 32 K-values are contiguous (64B), which is exactly
// the WMMA B-fragment access pattern (2x b128 per lane).
__global__ __launch_bounds__(256) void moe_cvt_w(const float* __restrict__ w,
                                                 unsigned short* __restrict__ wt,
                                                 int KD, int ND) {
    __shared__ __align__(16) unsigned short Lt[256][32];
    const int e = blockIdx.z, kt = blockIdx.y, n0 = blockIdx.x * 256;
    const int tid = threadIdx.x;
    const int kp = tid >> 4, c0 = (tid & 15) * 16;

    const float* r0 = w + ((size_t)e * KD + kt * 32 + 2 * kp) * ND + n0 + c0;
    const float* r1 = r0 + ND;
#pragma unroll
    for (int q = 0; q < 4; ++q) {
        float4 a = ((const float4*)r0)[q];
        float4 b = ((const float4*)r1)[q];
        *(uint32_t*)&Lt[c0 + 4 * q + 0][2 * kp] =
            (uint32_t)f2bf(a.x) | ((uint32_t)f2bf(b.x) << 16);
        *(uint32_t*)&Lt[c0 + 4 * q + 1][2 * kp] =
            (uint32_t)f2bf(a.y) | ((uint32_t)f2bf(b.y) << 16);
        *(uint32_t*)&Lt[c0 + 4 * q + 2][2 * kp] =
            (uint32_t)f2bf(a.z) | ((uint32_t)f2bf(b.z) << 16);
        *(uint32_t*)&Lt[c0 + 4 * q + 3][2 * kp] =
            (uint32_t)f2bf(a.w) | ((uint32_t)f2bf(b.w) << 16);
    }
    __syncthreads();
    unsigned short* dst = wt + ((size_t)e * (KD / 32) + kt) * ND * 32
                             + (size_t)(n0 + tid) * 32;
    const uint4* s4 = (const uint4*)&Lt[tid][0];
#pragma unroll
    for (int q = 0; q < 4; ++q) ((uint4*)dst)[q] = s4[q];
}

// --------------------------- router (one wave32 / token) -------------------
__global__ __launch_bounds__(256) void moe_router(
    const float* __restrict__ x, const float* __restrict__ gw,
    int* __restrict__ counts, int* __restrict__ tok_list,
    float* __restrict__ prob_list, float* __restrict__ probsum) {
    __shared__ float s_psum[E_];
    const int tid = threadIdx.x;
    if (tid < E_) s_psum[tid] = 0.f;
    __syncthreads();

    const int lane = tid & 31;
    const int t    = blockIdx.x * 8 + (tid >> 5);

    float acc[E_];
#pragma unroll
    for (int e = 0; e < E_; ++e) acc[e] = 0.f;

    const float* xr = x + (size_t)t * H_;
    for (int h = lane; h < H_; h += 32) {
        float xv = xr[h];
        const float4* g4 = (const float4*)(gw + (size_t)h * E_);
        float4 g0 = g4[0], g1 = g4[1];
        acc[0] += xv * g0.x; acc[1] += xv * g0.y;
        acc[2] += xv * g0.z; acc[3] += xv * g0.w;
        acc[4] += xv * g1.x; acc[5] += xv * g1.y;
        acc[6] += xv * g1.z; acc[7] += xv * g1.w;
    }
#pragma unroll
    for (int e = 0; e < E_; ++e)
#pragma unroll
        for (int off = 16; off > 0; off >>= 1)
            acc[e] += __shfl_xor(acc[e], off, 32);

    if (lane == 0) {
        float m = acc[0];
#pragma unroll
        for (int e = 1; e < E_; ++e) m = fmaxf(m, acc[e]);
        float p[E_], s = 0.f;
#pragma unroll
        for (int e = 0; e < E_; ++e) { p[e] = __expf(acc[e] - m); s += p[e]; }
        float inv = 1.f / s;
#pragma unroll
        for (int e = 0; e < E_; ++e) { p[e] *= inv; atomicAdd(&s_psum[e], p[e]); }
        int i0 = 0;
#pragma unroll
        for (int e = 1; e < E_; ++e) if (p[e] > p[i0]) i0 = e;
        int i1 = (i0 == 0) ? 1 : 0;
#pragma unroll
        for (int e = 0; e < E_; ++e) if (e != i0 && p[e] > p[i1]) i1 = e;
        float pn = 1.f / (p[i0] + p[i1]);
        int pos0 = atomicAdd(&counts[i0], 1);
        tok_list[i0 * T_ + pos0]  = t;
        prob_list[i0 * T_ + pos0] = p[i0] * pn;
        int pos1 = atomicAdd(&counts[i1], 1);
        tok_list[i1 * T_ + pos1]  = t;
        prob_list[i1 * T_ + pos1] = p[i1] * pn;
    }
    __syncthreads();
    if (tid < E_) atomicAdd(&probsum[tid], s_psum[tid]);
}

__global__ void moe_finalize(const int* __restrict__ counts,
                             const float* __restrict__ probsum,
                             int* __restrict__ base, float* __restrict__ aux_out) {
    if (threadIdx.x == 0) {
        int b = 0; float a = 0.f;
#pragma unroll
        for (int e = 0; e < E_; ++e) {
            base[e] = b; b += counts[e];
            float mean = probsum[e] / (float)T_;
            a += mean * mean;
        }
        *aux_out = (float)E_ * a;
    }
}

// --------------------------- fused GEMM kernel -----------------------------
// MODE 0 (fc1): A = gathered x rows,     D = relu(A@W + b1) -> hidden (bf16)
// MODE 1 (fc2): A = compact hidden rows, out[token] += prob*(A@W + b2) (f32)
//
// Block: 256 thr = 8 waves (4M x 2N), tile 128 x 256, K step 32.
// Wave tile 32 x 128: 2 A-frags, 8 B-frags, 16 WMMA (each B reused by 2 WMMA).
// A and B stream via global_load_async_to_lds_b128; frags = 2x ds_load_b128.
template <int MODE>
__global__ __launch_bounds__(256) void moe_gemm(
    const unsigned short* __restrict__ Ag,   // MODE0: [T][H]  MODE1: [RT][F]
    const unsigned short* __restrict__ Wg,   // tiled [E][KD/32][ND][32]
    const float* __restrict__ bias,          // [E][ND]
    const int* __restrict__ counts, const int* __restrict__ base,
    const int* __restrict__ tok_list, const float* __restrict__ prob_list,
    const unsigned short* __restrict__ zbuf, // >=32B of zeros
    unsigned short* __restrict__ hidden,     // MODE0 dest
    float* __restrict__ out) {               // MODE1 dest
    constexpr int KD = MODE ? F_ : H_;
    constexpr int ND = MODE ? H_ : F_;
    constexpr int NSTEP = KD / TK;

    const int e   = blockIdx.z;
    const int cnt = counts[e];
    const int m0  = blockIdx.y * TM;
    if (m0 >= cnt) return;
    const int n0  = blockIdx.x * TN;

    __shared__ __align__(16) unsigned short As[2][TM][LDK];  // 20.0 KB
    __shared__ __align__(16) unsigned short Bt[2][TN][LDK];  // 40.0 KB
    __shared__ int   s_tok[TM];
    __shared__ float s_prob[TM];

    const int tid = threadIdx.x;
    if (tid < TM) {
        int pos = m0 + tid;
        s_tok[tid]  = (pos < cnt) ? tok_list[e * T_ + pos] : -1;
        s_prob[tid] = (pos < cnt) ? prob_list[e * T_ + pos] : 0.f;
    }
    __syncthreads();

    const int lane = tid & 31, wave = tid >> 5;
    const int wm = wave >> 1, wn = wave & 1;   // 4 x 2 wave grid
    const int r15 = lane & 15, hh = lane >> 4;

    // staging: A = 16 bf16 / thread (one half-row), B = one 64B col / thread
    const int arow = tid >> 1, ah = (tid & 1) * 16;
    const int rbase = base[e];

    const unsigned short* asrc = zbuf;               // zeros for tail rows
    bool a_valid = false;
    if (MODE == 0) {
        int tok = s_tok[arow];
        if (tok >= 0) { asrc = Ag + (size_t)tok * KD + ah; a_valid = true; }
    } else {
        int pos = m0 + arow;
        if (pos < cnt) { asrc = Ag + (size_t)(rbase + pos) * KD + ah; a_valid = true; }
    }
    const unsigned short* btile = Wg + (size_t)e * (KD / 32) * ND * 32
                                     + (size_t)(n0 + tid) * 32;

    auto issue_stage = [&](int s, int buf) {
        const unsigned short* ap = a_valid ? (asrc + s * TK) : zbuf;
        async_copy16(ap,     &As[buf][arow][ah]);
        async_copy16(ap + 8, &As[buf][arow][ah + 8]);
        const unsigned short* bp = btile + (size_t)s * ((size_t)ND * 32);
#pragma unroll
        for (int q = 0; q < 4; ++q)
            async_copy16(bp + q * 8, &Bt[buf][tid][q * 8]);
        if (s + 1 < NSTEP)
            __builtin_prefetch(bp + (size_t)ND * 32, 0, 1);
    };

    v8f acc[2][8] = {};

    issue_stage(0, 0);
    wait_async0();
    __syncthreads();

    for (int s = 0; s < NSTEP; ++s) {
        const int cur = s & 1;
        const bool more = (s + 1 < NSTEP);
        if (more) issue_stage(s + 1, cur ^ 1);

        const unsigned short* pa0 = &As[cur][wm * 32 + r15][0];
        const unsigned short* pa1 = &As[cur][wm * 32 + 16 + r15][0];
        v16bf a0 = mk_frag(*(const uint4*)(pa0 + hh * 8),
                           *(const uint4*)(pa0 + 16 + hh * 8));
        v16bf a1 = mk_frag(*(const uint4*)(pa1 + hh * 8),
                           *(const uint4*)(pa1 + 16 + hh * 8));
#pragma unroll
        for (int st = 0; st < 8; ++st) {
            const unsigned short* pb = &Bt[cur][wn * 128 + st * 16 + r15][0];
            v16bf b = mk_frag(*(const uint4*)(pb + hh * 8),
                              *(const uint4*)(pb + 16 + hh * 8));
            acc[0][st] = __builtin_amdgcn_wmma_f32_16x16x32_bf16(
                false, a0, false, b, (short)0, acc[0][st], false, false);
            acc[1][st] = __builtin_amdgcn_wmma_f32_16x16x32_bf16(
                false, a1, false, b, (short)0, acc[1][st], false, false);
        }
        if (more) {
            wait_async0();
            __syncthreads();
        }
    }

    // epilogue: hoist per-lane bias values (independent of row)
    const float* bsrc = bias + (size_t)e * ND;
    float bv[8];
#pragma unroll
    for (int st = 0; st < 8; ++st)
        bv[st] = bsrc[n0 + wn * 128 + st * 16 + r15];

#pragma unroll
    for (int am = 0; am < 2; ++am) {
#pragma unroll
        for (int v = 0; v < 8; ++v) {
            const int rl  = wm * 32 + am * 16 + v + hh * 8;
            const int pos = m0 + rl;
            if (pos >= cnt) continue;
#pragma unroll
            for (int st = 0; st < 8; ++st) {
                const int c = n0 + wn * 128 + st * 16 + r15;
                float val = acc[am][st][v] + bv[st];
                if constexpr (MODE == 0) {
                    hidden[(size_t)(rbase + pos) * ND + c] = f2bf(fmaxf(val, 0.f));
                } else {
                    atomicAdd(&out[(size_t)s_tok[rl] * ND + c], s_prob[rl] * val);
                }
            }
        }
    }
}

// --------------------------- host launcher ---------------------------------
extern "C" void kernel_launch(void* const* d_in, const int* in_sizes, int n_in,
                              void* d_out, int out_size, void* d_ws, size_t ws_size,
                              hipStream_t stream) {
    (void)in_sizes; (void)n_in; (void)out_size; (void)ws_size;

    const float* x   = (const float*)d_in[0];  // [B,S,H]
    const float* gw  = (const float*)d_in[1];  // [H,E]
    const float* w1  = (const float*)d_in[2];  // [E,H,F]
    const float* b1  = (const float*)d_in[3];  // [E,F]
    const float* w2  = (const float*)d_in[4];  // [E,F,H]
    const float* b2  = (const float*)d_in[5];  // [E,H]
    float* out = (float*)d_out;                // [T*H] ++ [1] aux

    char* ws = (char*)d_ws;
    unsigned short* xb   = (unsigned short*)(ws + XB_OFF);
    unsigned short* w1t  = (unsigned short*)(ws + W1B_OFF);
    unsigned short* w2t  = (unsigned short*)(ws + W2B_OFF);
    unsigned short* hid  = (unsigned short*)(ws + HID_OFF);
    int*      tok   = (int*)(ws + TOK_OFF);
    float*    prb   = (float*)(ws + PRB_OFF);
    int*      cnt   = (int*)(ws + CNT_OFF);
    int*      basep = (int*)(ws + BAS_OFF);
    float*    psum  = (float*)(ws + PSM_OFF);
    uint32_t* zbuf  = (uint32_t*)(ws + ZB_OFF);

    moe_zero<<<2048, 256, 0, stream>>>(out, cnt, psum, zbuf);
    moe_cvt<<<256, 256, 0, stream>>>((const float4*)x, (ushort4*)xb,
                                     (size_t)T_ * H_ / 4);
    dim3 gw1(F_ / 256, H_ / 32, E_);   // (16, 32, 8)
    moe_cvt_w<<<gw1, 256, 0, stream>>>(w1, w1t, H_, F_);
    dim3 gw2(H_ / 256, F_ / 32, E_);   // (4, 128, 8)
    moe_cvt_w<<<gw2, 256, 0, stream>>>(w2, w2t, F_, H_);

    moe_router<<<T_ / 8, 256, 0, stream>>>(x, gw, cnt, tok, prb, psum);
    moe_finalize<<<1, 32, 0, stream>>>(cnt, psum, basep, out + (size_t)T_ * H_);

    dim3 g1(F_ / TN, T_ / TM, E_);   // (16, 32, 8)
    moe_gemm<0><<<g1, 256, 0, stream>>>(xb, w1t, b1, cnt, basep, tok, prb,
                                        (const unsigned short*)zbuf, hid, out);
    dim3 g2(H_ / TN, T_ / TM, E_);   // (4, 32, 8)
    moe_gemm<1><<<g2, 256, 0, stream>>>(hid, w2t, b2, cnt, basep, tok, prb,
                                        (const unsigned short*)zbuf, hid, out);
}
